// Matrix_Decoder_with_critic_39608188404204
// MI455X (gfx1250) — compile-verified, matrix operand order checked by reading, not run on hardware
//
#include <hip/hip_runtime.h>
#include <math.h>

typedef __attribute__((ext_vector_type(16))) _Float16 v16h;
typedef __attribute__((ext_vector_type(4)))  _Float16 v4h;
typedef __attribute__((ext_vector_type(8)))  float    v8f;
typedef __attribute__((ext_vector_type(4)))  float    v4f;

#define B_    256
#define M_    5
#define T_    2048
#define EMB_  128
#define H_    8
#define DQK_  16
#define TT    128          // t-tile rows for kernel 1
#define NT    (T_/TT)      // 16 tiles
#define TT2   256          // t-tile rows for kernel 2
#define NT2   (T_/TT2)     // 8 tiles
#define FPAD  1            // float-element pad (bank-conflict avoidance)

// Fragment-order layout for WMMA operands:
//   [ct|rowtile][chunk c][quad q][lane][4 halfs], fragment elem idx = q*4+h.
// ISA 16-bit 16x32 layout: lane = g*16 + (m|n), elem idx = hi*8 + e with
//   K = c*32 + hi*16 + g*8 + e.

// -------------------------------------------------------------------------
// Kernel 1: fused QKV projection (f16 WMMA) + flattened-softmax attention +
//           Wc projection + critic head. One workgroup per batch element.
// -------------------------------------------------------------------------
__global__ __launch_bounds__(256) void k1_attention(
    const float* __restrict__ machine,   // [B,5,128]
    const float* __restrict__ jobs,      // [B,2048,128]
    const float* __restrict__ mask,      // [B,5,2048]
    const float* __restrict__ Wq3,       // [128,128]
    const float* __restrict__ Wk,        // [128,128]
    const float* __restrict__ Wv,        // [128,128]
    const float* __restrict__ Wc,        // [128,128]
    const float* __restrict__ bc,        // [128]
    const float* __restrict__ Wg,        // [128]
    const float* __restrict__ bg,        // [1]
    float* __restrict__ mh_ws,           // [B,5,128] scratch
    float* __restrict__ gt_out)          // [B]
{
    __shared__ _Float16 fWk[8][4][4][32][4];   // 32 KB, B fragments of Wk
    __shared__ _Float16 fWv[8][4][4][32][4];   // 32 KB, B fragments of Wv
    __shared__ _Float16 fA [8][4][4][32][4];   // 32 KB, A fragments of jobs tile
    __shared__ float    sK [TT][EMB_ + FPAD];  // 66 KB
    __shared__ float    sV [TT][EMB_ + FPAD];  // 66 KB
    __shared__ float    sQ [M_][EMB_];
    __shared__ float    sOC[M_][EMB_];
    __shared__ float    sMH[M_][EMB_];

    const int b    = blockIdx.x;
    const int tid  = threadIdx.x;
    const int w    = tid >> 5;       // wave id == head id (8 waves, H=8)
    const int lane = tid & 31;
    const int g    = lane >> 4;      // lane group (WMMA C layout)
    const int mr   = lane & 15;

    // ---- stage f16 weights in LDS in fragment order (once) ----
    for (int i = tid; i < EMB_ * EMB_; i += 256) {
        const int k  = i >> 7, n = i & 127;
        const int ct = n >> 4, nn = n & 15;
        const int c  = k >> 5, off = k & 31;
        const int hi = off >> 4, r = off & 15;
        const int gg = r >> 3,  e = r & 7;
        const int idx = hi * 8 + e;
        fWk[ct][c][idx >> 2][gg * 16 + nn][idx & 3] = (_Float16)Wk[i];
        fWv[ct][c][idx >> 2][gg * 16 + nn][idx & 3] = (_Float16)Wv[i];
    }
    // ---- q = machine[b] @ Wq3  (tiny, fp32 VALU) ----
    for (int i = tid; i < M_ * EMB_; i += 256) {
        int m = i >> 7, c = i & 127;
        const float* mrow = machine + ((size_t)b * M_ + m) * EMB_;
        float acc = 0.f;
        for (int e = 0; e < EMB_; ++e) acc += mrow[e] * Wq3[e * EMB_ + c];
        sQ[m][c] = acc;
    }
    __syncthreads();

    // per-lane online-softmax state for head w (softmax over flattened m*t)
    float run_max = -INFINITY;
    float run_sum = 0.f;
    float oacc[M_][DQK_];
    #pragma unroll
    for (int m = 0; m < M_; ++m)
        #pragma unroll
        for (int d = 0; d < DQK_; ++d) oacc[m][d] = 0.f;

    for (int ti = 0; ti < NT; ++ti) {
        const int t0 = ti * TT;

        // ---- load jobs tile, convert fp32->f16, store in A-fragment order ----
        const float* jb = jobs + ((size_t)b * T_ + t0) * EMB_;
        for (int i = tid; i < TT * EMB_ / 4; i += 256) {
            v4f x = ((const v4f*)jb)[i];
            const int base = i * 4;
            const int row = base >> 7, kk = base & 127;
            const int wt = row >> 4,  rm = row & 15;
            const int c  = kk >> 5,  off = kk & 31;
            const int hi = off >> 4, r = off & 15;
            const int gg = r >> 3;               // e = r&7 in {0,4}
            const int idx = hi * 8 + (r & 7);    // multiple of 4
            v4h h;
            h[0] = (_Float16)x.x; h[1] = (_Float16)x.y;
            h[2] = (_Float16)x.z; h[3] = (_Float16)x.w;
            *(v4h*)&fA[wt][c][idx >> 2][gg * 16 + rm][0] = h;
        }
        __syncthreads();

        // ---- k/v projection via v_wmma_f32_16x16x32_f16 ----
        // wave w computes output rows t = w*16 .. w*16+15, all 128 cols.
        v16h A[4];
        #pragma unroll
        for (int c = 0; c < 4; ++c) {
            union { v16h v; v4h q[4]; } u;
            #pragma unroll
            for (int qq = 0; qq < 4; ++qq)
                u.q[qq] = *(const v4h*)&fA[w][c][qq][lane][0];
            A[c] = u.v;
        }
        #pragma unroll
        for (int ct = 0; ct < 8; ++ct) {
            v8f accK = {};
            v8f accV = {};
            #pragma unroll
            for (int c = 0; c < 4; ++c) {
                union { v16h v; v4h q[4]; } uk, uv;
                #pragma unroll
                for (int qq = 0; qq < 4; ++qq) {
                    uk.q[qq] = *(const v4h*)&fWk[ct][c][qq][lane][0];
                    uv.q[qq] = *(const v4h*)&fWv[ct][c][qq][lane][0];
                }
                accK = __builtin_amdgcn_wmma_f32_16x16x32_f16(
                           false, A[c], false, uk.v, (short)0, accK, false, false);
                accV = __builtin_amdgcn_wmma_f32_16x16x32_f16(
                           false, A[c], false, uv.v, (short)0, accV, false, false);
            }
            const int col = ct * 16 + mr;
            #pragma unroll
            for (int r = 0; r < 8; ++r) {        // C layout: M = r + g*8
                sK[w*16 + r + g*8][col] = accK[r];
                sV[w*16 + r + g*8][col] = accV[r];
            }
        }
        __syncthreads();

        // ---- attention for head w over this tile (online softmax) ----
        float ts[4][M_];
        float tmax = -INFINITY;
        #pragma unroll
        for (int i = 0; i < 4; ++i) {
            const int tl = lane + 32 * i;
            float kreg[DQK_];
            #pragma unroll
            for (int d = 0; d < DQK_; ++d) kreg[d] = sK[tl][w*16 + d];
            #pragma unroll
            for (int m = 0; m < M_; ++m) {
                float s = 0.f;
                #pragma unroll
                for (int d = 0; d < DQK_; ++d) s += sQ[m][w*16 + d] * kreg[d];
                s = s * 0.25f + mask[((size_t)b * M_ + m) * T_ + t0 + tl];
                ts[i][m] = s;
                tmax = fmaxf(tmax, s);
            }
        }
        const float nm    = fmaxf(run_max, tmax);
        const float scale = __expf(run_max - nm);
        run_sum *= scale;
        #pragma unroll
        for (int m = 0; m < M_; ++m)
            #pragma unroll
            for (int d = 0; d < DQK_; ++d) oacc[m][d] *= scale;
        #pragma unroll
        for (int i = 0; i < 4; ++i) {
            const int tl = lane + 32 * i;
            float vreg[DQK_];
            #pragma unroll
            for (int d = 0; d < DQK_; ++d) vreg[d] = sV[tl][w*16 + d];
            #pragma unroll
            for (int m = 0; m < M_; ++m) {
                const float wt = __expf(ts[i][m] - nm);
                run_sum += wt;
                #pragma unroll
                for (int d = 0; d < DQK_; ++d) oacc[m][d] += wt * vreg[d];
            }
        }
        run_max = nm;
        __syncthreads();
    }

    // ---- merge softmax state across the 32 lanes (butterfly) ----
    #pragma unroll
    for (int off = 16; off >= 1; off >>= 1) {
        const float om = __shfl_xor(run_max, off, 32);
        const float os = __shfl_xor(run_sum, off, 32);
        const float nm = fmaxf(run_max, om);
        const float e1 = __expf(run_max - nm);
        const float e2 = __expf(om - nm);
        run_sum = run_sum * e1 + os * e2;
        #pragma unroll
        for (int m = 0; m < M_; ++m)
            #pragma unroll
            for (int d = 0; d < DQK_; ++d) {
                const float ov = __shfl_xor(oacc[m][d], off, 32);
                oacc[m][d] = oacc[m][d] * e1 + ov * e2;
            }
        run_max = nm;
    }
    const float inv = 1.0f / run_sum;
    // all lanes now hold identical state; same-value LDS writes are benign
    #pragma unroll
    for (int m = 0; m < M_; ++m)
        #pragma unroll
        for (int d = 0; d < DQK_; ++d)
            sOC[m][w*16 + d] = oacc[m][d] * inv;
    __syncthreads();

    // ---- mh = out_concat @ Wc + bc ----
    for (int i = tid; i < M_ * EMB_; i += 256) {
        int m = i >> 7, o = i & 127;
        float acc = bc[o];
        for (int e = 0; e < EMB_; ++e) acc += sOC[m][e] * Wc[e * EMB_ + o];
        sMH[m][o] = acc;
        mh_ws[((size_t)b * M_ + m) * EMB_ + o] = acc;
    }
    __syncthreads();

    // ---- critic: G_t = max_m(mh) . Wg + bg  (wave 0) ----
    if (w == 0) {
        float part = 0.f;
        for (int o = lane; o < EMB_; o += 32) {
            float p = sMH[0][o];
            #pragma unroll
            for (int m = 1; m < M_; ++m) p = fmaxf(p, sMH[m][o]);
            part += p * Wg[o];
        }
        #pragma unroll
        for (int off = 16; off >= 1; off >>= 1)
            part += __shfl_xor(part, off, 32);
        if (lane == 0) gt_out[b] = part + bg[0];
    }
}

// -------------------------------------------------------------------------
// Kernel 2: score2 = 10*tanh(mh @ jobs^T / sqrt(128)) + mask, softmax over
//           flattened M*T per batch. One workgroup per batch element.
// -------------------------------------------------------------------------
__global__ __launch_bounds__(256) void k2_probs(
    const float* __restrict__ jobs,      // [B,2048,128]
    const float* __restrict__ mask,      // [B,5,2048]
    const float* __restrict__ mh_ws,     // [B,5,128]
    float* __restrict__ probs)           // [B, M*T]
{
    __shared__ float sJ[TT2][EMB_ + FPAD];
    __shared__ float sMH[M_][EMB_];
    __shared__ float sRmax[8];
    __shared__ float sRsum[8];

    const int b   = blockIdx.x;
    const int tid = threadIdx.x;

    for (int i = tid; i < M_ * EMB_; i += 256)
        ((float*)sMH)[ (i >> 7) * EMB_ + (i & 127) ] = mh_ws[(size_t)b * M_ * EMB_ + i];
    __syncthreads();

    float sc[M_][NT2];                   // 40 registers of scores
    #pragma unroll
    for (int ti = 0; ti < NT2; ++ti) {
        const float* jb = jobs + ((size_t)b * T_ + ti * TT2) * EMB_;
        for (int i = tid; i < TT2 * EMB_ / 4; i += 256) {
            v4f x = ((const v4f*)jb)[i];
            int base = i * 4;
            int row = base >> 7, col = base & 127;
            sJ[row][col + 0] = x.x;
            sJ[row][col + 1] = x.y;
            sJ[row][col + 2] = x.z;
            sJ[row][col + 3] = x.w;
        }
        __syncthreads();
        #pragma unroll
        for (int m = 0; m < M_; ++m) {
            float acc = 0.f;
            #pragma unroll 8
            for (int e = 0; e < EMB_; ++e) acc += sMH[m][e] * sJ[tid][e];
            acc = 10.0f * tanhf(acc * 0.08838834764831845f)   // 1/sqrt(128)
                  + mask[((size_t)b * M_ + m) * T_ + ti * TT2 + tid];
            sc[m][ti] = acc;
        }
        __syncthreads();
    }

    // block-wide max
    float lmax = -INFINITY;
    #pragma unroll
    for (int m = 0; m < M_; ++m)
        #pragma unroll
        for (int ti = 0; ti < NT2; ++ti) lmax = fmaxf(lmax, sc[m][ti]);
    #pragma unroll
    for (int off = 16; off >= 1; off >>= 1)
        lmax = fmaxf(lmax, __shfl_xor(lmax, off, 32));
    if ((tid & 31) == 0) sRmax[tid >> 5] = lmax;
    __syncthreads();
    float gmax = sRmax[0];
    #pragma unroll
    for (int i = 1; i < 8; ++i) gmax = fmaxf(gmax, sRmax[i]);

    // block-wide exp-sum
    float lsum = 0.f;
    #pragma unroll
    for (int m = 0; m < M_; ++m)
        #pragma unroll
        for (int ti = 0; ti < NT2; ++ti) {
            const float e = __expf(sc[m][ti] - gmax);
            sc[m][ti] = e;
            lsum += e;
        }
    #pragma unroll
    for (int off = 16; off >= 1; off >>= 1)
        lsum += __shfl_xor(lsum, off, 32);
    if ((tid & 31) == 0) sRsum[tid >> 5] = lsum;
    __syncthreads();
    float gsum = 0.f;
    #pragma unroll
    for (int i = 0; i < 8; ++i) gsum += sRsum[i];
    const float inv = 1.0f / gsum;

    // coalesced writes: index = b*(M*T) + m*T + ti*TT2 + tid
    float* po = probs + (size_t)b * (M_ * T_);
    #pragma unroll
    for (int m = 0; m < M_; ++m)
        #pragma unroll
        for (int ti = 0; ti < NT2; ++ti)
            po[m * T_ + ti * TT2 + tid] = sc[m][ti] * inv;
}

// -------------------------------------------------------------------------
extern "C" void kernel_launch(void* const* d_in, const int* in_sizes, int n_in,
                              void* d_out, int out_size, void* d_ws, size_t ws_size,
                              hipStream_t stream) {
    const float* machine = (const float*)d_in[0];
    const float* jobs    = (const float*)d_in[1];
    const float* mask    = (const float*)d_in[2];
    const float* Wq3     = (const float*)d_in[3];
    const float* Wk      = (const float*)d_in[4];
    const float* Wv      = (const float*)d_in[5];
    const float* Wc      = (const float*)d_in[6];
    const float* bc      = (const float*)d_in[7];
    const float* Wg      = (const float*)d_in[8];
    const float* bg      = (const float*)d_in[9];

    float* probs = (float*)d_out;                      // B*M*T floats
    float* gt    = probs + (size_t)B_ * M_ * T_;       // B floats
    float* mh_ws = (float*)d_ws;                       // B*M*EMB floats

    k1_attention<<<B_, 256, 0, stream>>>(machine, jobs, mask, Wq3, Wk, Wv,
                                         Wc, bc, Wg, bg, mh_ws, gt);
    k2_probs<<<B_, 256, 0, stream>>>(jobs, mask, mh_ws, probs);
}
